// ScaledDotProductAttention_26628797235623
// MI455X (gfx1250) — matrix operand order repaired and checked by reading, not compiled, not run on hardware
//
#include <hip/hip_runtime.h>
#include <hip/hip_bf16.h>

typedef __attribute__((ext_vector_type(16))) _Float16 v16h;
typedef __attribute__((ext_vector_type(8)))  float    v8f;

#define B_   8
#define C_   64
#define HW_  4096          // 64*64
#define CR_  16            // reduction channels
#define GRP_ 4             // groups
#define GC_  16            // group channels
#define KK_  49            // 7*7
#define OSP_ 196           // K*K*G = span output channels
#define OSPP_ 208          // OSP padded to 13*16
#define NPX_ (B_ * HW_)    // 32768 pixels

// A-fragment K index (contiguous order, matching B):
// lanes 0-15 hold K in {0..15}; lanes 16-31 hold K in {16..31}
__device__ __forceinline__ int a_kidx(int i, int hl) {
    return 2 * hl + 2 * i;
}

__device__ __forceinline__ v8f wmma_f16(v16h a, v16h b, v8f c) {
    return __builtin_amdgcn_wmma_f32_16x16x32_f16(false, a, false, b,
                                                  (short)0, c, false, false);
}

// ---------------------------------------------------------------- kernel 0
__global__ void inv_init_stats(float* __restrict__ stats) {
    if (threadIdx.x < 64) stats[threadIdx.x] = 0.0f;
}

// ---------------------------------------------------------------- kernel 1
// r = w1 (16x64) * x (64 x 32768), WMMA 16x16x32_f16, K=64 via 2 steps.
// Each wave: 4 consecutive N-tiles (64 pixels), A loaded once (4x reuse),
// 8 WMMAs per wave. Also accumulates per-channel sum/sumsq for BN.
__global__ void inv_reduce_gemm_stats(const float* __restrict__ x,    // [8][64][4096]
                                      const float* __restrict__ w1,   // [16][64]
                                      float* __restrict__ r_raw,      // [8][16][4096]
                                      float* __restrict__ stats) {    // [64]
    __shared__ float s_sum[CR_];
    __shared__ float s_ssq[CR_];
    const int tid = threadIdx.x;
    if (tid < CR_) { s_sum[tid] = 0.0f; s_ssq[tid] = 0.0f; }
    __syncthreads();

    const int lane = tid & 31;
    const int wave = tid >> 5;
    const int tg   = blockIdx.x * 8 + wave;         // [0, 512) exact
    const int p0   = tg * 64;                       // 64 pixels per wave
    const int b    = p0 >> 12;
    const int hw0  = p0 & 4095;

    const int m    = lane & 15;                     // A row / D row base
    const int n    = lane & 15;                     // B/D column
    const int hl   = (lane < 16) ? 0 : 8;           // A K-half select
    const int kh16 = (lane < 16) ? 0 : 16;          // B K-half select

    // A fragments: w1 rows, two K-chunks of 32 (loaded once, reused 4x)
    v16h a0, a1;
    #pragma unroll
    for (int i = 0; i < 8; ++i) {
        const int kb = a_kidx(i, hl);
        a0[2 * i]     = (_Float16)w1[m * 64 + kb];
        a0[2 * i + 1] = (_Float16)w1[m * 64 + kb + 1];
        a1[2 * i]     = (_Float16)w1[m * 64 + 32 + kb];
        a1[2 * i + 1] = (_Float16)w1[m * 64 + 32 + kb + 1];
    }

    const float* xb = x + (size_t)b * (C_ * HW_) + hw0 + n;
    float*       rb = r_raw + (size_t)b * (CR_ * HW_) + hw0 + n;
    __builtin_prefetch(xb + 64, 0, 0);              // next wave-tile -> global_prefetch_b8

    #pragma unroll
    for (int s4 = 0; s4 < 4; ++s4) {
        const float* xs = xb + s4 * 16;
        v16h b0, b1;
        #pragma unroll
        for (int j = 0; j < 8; ++j) {
            const int k = kh16 + 2 * j;
            b0[2 * j]     = (_Float16)xs[(size_t)k * HW_];
            b0[2 * j + 1] = (_Float16)xs[(size_t)(k + 1) * HW_];
            b1[2 * j]     = (_Float16)xs[(size_t)(k + 32) * HW_];
            b1[2 * j + 1] = (_Float16)xs[(size_t)(k + 33) * HW_];
        }

        v8f acc = {};
        acc = wmma_f16(a0, b0, acc);
        acc = wmma_f16(a1, b1, acc);

        // D layout: VGPR v holds rows v (lanes 0-15) / v+8 (lanes 16-31)
        float* rs = rb + s4 * 16;
        #pragma unroll
        for (int v = 0; v < 8; ++v) {
            const int mm = v + hl;
            const float val = acc[v];
            rs[(size_t)mm * HW_] = val;
            atomicAdd(&s_sum[mm], val);        // ds_add_f32
            atomicAdd(&s_ssq[mm], val * val);
        }
    }
    __syncthreads();
    if (tid < CR_) {
        atomicAdd(&stats[tid], s_sum[tid]);        // global_atomic_add_f32
        atomicAdd(&stats[CR_ + tid], s_ssq[tid]);
    }
}

// ---------------------------------------------------------------- kernel 2
// stats[32..47] = scale = gamma*rsqrt(var+eps);  stats[48..63] = shift
__global__ void inv_bn_finalize(const float* __restrict__ gamma,
                                const float* __restrict__ beta,
                                float* __restrict__ stats) {
    const int t = threadIdx.x;
    if (t < CR_) {
        const float invN = 1.0f / (float)NPX_;
        const float mean = stats[t] * invN;
        const float var  = stats[CR_ + t] * invN - mean * mean;
        const float sc   = gamma[t] * rsqrtf(var + 1e-5f);
        stats[32 + t] = sc;
        stats[48 + t] = beta[t] - mean * sc;
    }
}

// ---------------------------------------------------------------- kernel 3
// wgen = w2 (196x16) * relu(bn(r)) (16 x 32768) + b2.
// One wave per 16-pixel N-tile: B built once (fused BN+ReLU), then all
// 13 M-tiles computed from it (13x B reuse, 13 WMMAs). w2/b2 staged in
// LDS as f16/f32, zero-padded to 208 rows, so A builds are ds loads.
__global__ void inv_span_gemm(const float* __restrict__ r_raw,  // [8][16][4096]
                              const float* __restrict__ w2,     // [196][16]
                              const float* __restrict__ b2,     // [196]
                              const float* __restrict__ stats,
                              float* __restrict__ wgen) {       // [8][196][4096]
    __shared__ _Float16 sw2[OSPP_ * CR_];
    __shared__ float    sb2[OSPP_];

    const int tid = threadIdx.x;
    for (int i = tid; i < OSPP_ * CR_; i += 256) {
        const int row = i >> 4;
        sw2[i] = (row < OSP_) ? (_Float16)w2[i] : (_Float16)0.0f;
    }
    for (int i = tid; i < OSPP_; i += 256)
        sb2[i] = (i < OSP_) ? b2[i] : 0.0f;
    __syncthreads();

    const int lane = tid & 31;
    const int wave = tid >> 5;
    const int pt   = blockIdx.x * 8 + wave;         // [0, 2048) exact
    const int p0   = pt * 16;
    const int b    = p0 >> 12;
    const int hw0  = p0 & 4095;

    const int m    = lane & 15;
    const int n    = lane & 15;
    const int hl   = (lane < 16) ? 0 : 8;
    const int kh16 = (lane < 16) ? 0 : 16;

    // B built once: fused BN + ReLU on r, zero-pad K>=16
    const float* rb = r_raw + (size_t)b * (CR_ * HW_) + hw0 + n;
    v16h bf = {};
    #pragma unroll
    for (int j = 0; j < 8; ++j) {
        #pragma unroll
        for (int s = 0; s < 2; ++s) {
            const int k = kh16 + 2 * j + s;
            _Float16 val = (_Float16)0.0f;
            if (k < CR_) {
                float rv = rb[(size_t)k * HW_];
                rv = fmaxf(rv * stats[32 + k] + stats[48 + k], 0.0f);
                val = (_Float16)rv;
            }
            bf[2 * j + s] = val;
        }
    }

    float* wo = wgen + (size_t)b * (OSP_ * HW_) + hw0 + n;

    for (int mt = 0; mt < 13; ++mt) {
        // A from LDS (zero-pad K>=16; rows >=196 already zeroed in LDS)
        const int mg_a = mt * 16 + m;
        v16h a = {};
        #pragma unroll
        for (int i = 0; i < 8; ++i) {
            const int kb = a_kidx(i, hl);
            #pragma unroll
            for (int s = 0; s < 2; ++s) {
                const int k = kb + s;
                a[2 * i + s] = (k < CR_) ? sw2[mg_a * CR_ + k] : (_Float16)0.0f;
            }
        }

        // C: bias broadcast along N (sb2 zero-padded -> no guard)
        v8f acc;
        #pragma unroll
        for (int v = 0; v < 8; ++v)
            acc[v] = sb2[mt * 16 + v + hl];

        acc = wmma_f16(a, bf, acc);

        #pragma unroll
        for (int v = 0; v < 8; ++v) {
            const int mg = mt * 16 + v + hl;
            if (mg < OSP_) wo[(size_t)mg * HW_] = acc[v];
        }
    }
}

// ---------------------------------------------------------------- kernel 4
// Temporal diffs (6 ch) + 7x7 SAME conv (6->4).  Writes d_out (=).
// grid (4,4,128), block (16,16). z = n = b*16 + cgrp; data chans cgrp*4 + t.
__global__ void tdiff_conv7(const float* __restrict__ x,      // [8][64][4096]
                            const float* __restrict__ w,      // [4][6][7][7]
                            float* __restrict__ out) {        // [8][64][4096]
    __shared__ float sd[4][22][22];
    __shared__ float sw[4 * 6 * 49];

    const int tx = threadIdx.x, ty = threadIdx.y;
    const int tid = ty * 16 + tx;
    const int nz = blockIdx.z;
    const int b = nz >> 4;
    const int cbase = (nz & 15) * 4;
    const int h0 = blockIdx.y * 16 - 3;
    const int w0 = blockIdx.x * 16 - 3;

    for (int i = tid; i < 4 * 6 * 49; i += 256) sw[i] = w[i];

    const float* xb = x + (size_t)b * (C_ * HW_) + (size_t)cbase * HW_;
    #pragma unroll
    for (int c = 0; c < 4; ++c) {
        for (int i = tid; i < 22 * 22; i += 256) {
            const int iy = i / 22, ix = i % 22;
            const int gh = h0 + iy, gw = w0 + ix;
            float v = 0.0f;
            if (gh >= 0 && gh < 64 && gw >= 0 && gw < 64)
                v = xb[(size_t)c * HW_ + gh * 64 + gw];
            sd[c][iy][ix] = v;
        }
    }
    __syncthreads();

    float acc[4] = {0.0f, 0.0f, 0.0f, 0.0f};
    #pragma unroll
    for (int kh = 0; kh < 7; ++kh) {
        #pragma unroll
        for (int kw = 0; kw < 7; ++kw) {
            const float d0 = sd[0][ty + kh][tx + kw];
            const float d1 = sd[1][ty + kh][tx + kw];
            const float d2 = sd[2][ty + kh][tx + kw];
            const float d3 = sd[3][ty + kh][tx + kw];
            // diff channel order: d0-d1, d1-d2, d2-d3, d0-d2, d1-d3, d0-d3
            float df[6];
            df[0] = d0 - d1; df[1] = d1 - d2; df[2] = d2 - d3;
            df[3] = d0 - d2; df[4] = d1 - d3; df[5] = d0 - d3;
            const int kofs = kh * 7 + kw;
            #pragma unroll
            for (int to = 0; to < 4; ++to)
                #pragma unroll
                for (int ci = 0; ci < 6; ++ci)
                    acc[to] = fmaf(sw[(to * 6 + ci) * 49 + kofs], df[ci], acc[to]);
        }
    }

    const int hh = blockIdx.y * 16 + ty;
    const int ww = blockIdx.x * 16 + tx;
    float* ob = out + (size_t)b * (C_ * HW_) + hh * 64 + ww;
    #pragma unroll
    for (int to = 0; to < 4; ++to)
        ob[(size_t)(cbase + to) * HW_] = acc[to];
}

// ---------------------------------------------------------------- kernel 5
// Involution aggregation: out += sum_k x[g*16+c, p+off(k)] * wgen[g,k,p].
// grid (4,4,32) z = b*4+g, block (16,16). 16-channel halo tile in LDS.
__global__ void inv_apply(const float* __restrict__ x,       // [8][64][4096]
                          const float* __restrict__ wgen,    // [8][196][4096]
                          float* __restrict__ out) {         // [8][64][4096]
    __shared__ float xs[GC_][22][22];

    const int tx = threadIdx.x, ty = threadIdx.y;
    const int tid = ty * 16 + tx;
    const int z = blockIdx.z;
    const int b = z >> 2;
    const int g = z & 3;
    const int h0 = blockIdx.y * 16 - 3;
    const int w0 = blockIdx.x * 16 - 3;

    const float* xb = x + (size_t)b * (C_ * HW_) + (size_t)(g * GC_) * HW_;
    for (int i = tid; i < GC_ * 22 * 22; i += 256) {
        const int c = i / (22 * 22);
        const int r = i % (22 * 22);
        const int iy = r / 22, ix = r % 22;
        const int gh = h0 + iy, gw = w0 + ix;
        float v = 0.0f;
        if (gh >= 0 && gh < 64 && gw >= 0 && gw < 64)
            v = xb[(size_t)c * HW_ + gh * 64 + gw];
        xs[c][iy][ix] = v;
    }
    __syncthreads();

    const int hh = blockIdx.y * 16 + ty;
    const int ww = blockIdx.x * 16 + tx;
    const int pix = hh * 64 + ww;
    const float* wg = wgen + (size_t)b * (OSP_ * HW_) + (size_t)(g * KK_) * HW_ + pix;

    float acc[GC_];
    #pragma unroll
    for (int c = 0; c < GC_; ++c) acc[c] = 0.0f;

    #pragma unroll
    for (int k = 0; k < KK_; ++k) {
        const int kh = k / 7, kw = k % 7;
        const float wk = wg[(size_t)k * HW_];     // coalesced across lanes
        #pragma unroll
        for (int c = 0; c < GC_; ++c)
            acc[c] = fmaf(xs[c][ty + kh][tx + kw], wk, acc[c]);
    }

    float* ob = out + (size_t)b * (C_ * HW_) + (size_t)(g * GC_) * HW_ + pix;
    #pragma unroll
    for (int c = 0; c < GC_; ++c)
        ob[(size_t)c * HW_] += acc[c];
}

// ---------------------------------------------------------------- launch
extern "C" void kernel_launch(void* const* d_in, const int* in_sizes, int n_in,
                              void* d_out, int out_size, void* d_ws, size_t ws_size,
                              hipStream_t stream) {
    const float* data     = (const float*)d_in[0];
    const float* conv1_w  = (const float*)d_in[1];
    const float* inv_w1   = (const float*)d_in[2];
    const float* bn_gamma = (const float*)d_in[3];
    const float* bn_beta  = (const float*)d_in[4];
    const float* inv_w2   = (const float*)d_in[5];
    const float* inv_b2   = (const float*)d_in[6];
    float* out = (float*)d_out;

    float* ws    = (float*)d_ws;
    float* stats = ws;                                 // 256 floats
    float* r_raw = ws + 256;                           // 8*16*4096
    float* wgen  = r_raw + (size_t)B_ * CR_ * HW_;     // 8*196*4096

    inv_init_stats<<<1, 64, 0, stream>>>(stats);
    inv_reduce_gemm_stats<<<64, 256, 0, stream>>>(data, inv_w1, r_raw, stats);
    inv_bn_finalize<<<1, 32, 0, stream>>>(bn_gamma, bn_beta, stats);
    inv_span_gemm<<<256, 256, 0, stream>>>(r_raw, inv_w2, inv_b2, stats, wgen);

    dim3 blk(16, 16, 1);
    tdiff_conv7<<<dim3(4, 4, 128), blk, 0, stream>>>(data, conv1_w, out);
    inv_apply<<<dim3(4, 4, 32), blk, 0, stream>>>(data, wgen, out);
}
